// LogFF_712964571628
// MI455X (gfx1250) — matrix operand order repaired and checked by reading, not compiled, not run on hardware
//
#include <hip/hip_runtime.h>

// ---------- CDNA5 async global->LDS path (guarded) ----------
#if defined(__has_builtin)
# if __has_builtin(__builtin_amdgcn_global_load_async_to_lds_b128) && \
     __has_builtin(__builtin_amdgcn_s_wait_asynccnt)
#  define USE_ASYNC 1
# endif
#endif
#ifndef USE_ASYNC
# define USE_ASYNC 0
#endif

// ---------- types ----------
typedef __bf16 bf16;
typedef bf16 v16bf __attribute__((ext_vector_type(16)));
typedef float v8f  __attribute__((ext_vector_type(8)));
typedef unsigned int uu4 __attribute__((ext_vector_type(4)));

union Frag16 { v16bf v; uu4 q[2]; };

#define B_TOT    8192
#define D_IN     1024
#define D_OUT    1024
#define LEAF     64
#define N_LEAVES 128
#define N_NODES  127
#define KFLAT    (N_LEAVES * LEAF)   // 8192
#define KSTEP    64

#if USE_ASYNC
// builtin signature (from hipcc diagnostic): (int4 __device__*, int4 __shared__*, Ii, Ii)
typedef int v4i __attribute__((ext_vector_type(4)));
typedef __attribute__((address_space(1))) v4i g_v4i;
typedef __attribute__((address_space(3))) v4i l_v4i;
__device__ __forceinline__ g_v4i* glb_cast(const void* p) {
  return (g_v4i*)(unsigned long long)p;          // generic global ptr value == global address
}
__device__ __forceinline__ l_v4i* lds_cast(const void* p) {
  return (l_v4i*)(unsigned int)(unsigned long long)p; // low 32 bits of generic LDS ptr == LDS offset
}
#endif

__device__ __forceinline__ unsigned short f32_to_bf16(float f) {
  unsigned int u = __float_as_uint(f);
  u += 0x7FFFu + ((u >> 16) & 1u);   // round-to-nearest-even
  return (unsigned short)(u >> 16);
}

__device__ __forceinline__ v8f wmma_bf16(const Frag16& a, const Frag16& b, v8f c) {
  return __builtin_amdgcn_wmma_f32_16x16x32_bf16(false, a.v, false, b.v, (short)0, c, false, false);
}

// ---------- x: fp32 -> bf16 (row-major, unchanged layout) ----------
__global__ void conv_f32_bf16(const float* __restrict__ src,
                              unsigned short* __restrict__ dst, int n) {
  int i = blockIdx.x * blockDim.x + threadIdx.x;
  int stride = gridDim.x * blockDim.x;
  for (; i < n; i += stride) dst[i] = f32_to_bf16(src[i]);
}

// ---------- w1s[l][i][j] -> w1t[(l*64+j)][i]  (bf16, K-contiguous per output col) ----------
__global__ void __launch_bounds__(256) transpose_w1(const float* __restrict__ src,
                                                    unsigned short* __restrict__ dst) {
  const int idx = blockIdx.x * 256 + threadIdx.x;  // 128*64*1024 total
  const int l = idx >> 16;
  const int j = (idx >> 10) & 63;
  const int i = idx & 1023;
  dst[idx] = f32_to_bf16(src[((size_t)l << 16) + i * 64 + j]);
}

// ---------- w2s[(l*64+j)][o] -> w2t[o][(l*64+j)]  (bf16, K-contiguous per output col) ----------
__global__ void __launch_bounds__(256) transpose_w2(const float* __restrict__ src,
                                                    unsigned short* __restrict__ dst) {
  const int idx = blockIdx.x * 256 + threadIdx.x;  // 1024*8192 total
  const int o = idx >> 13;
  const int r = idx & 8191;
  dst[idx] = f32_to_bf16(src[(size_t)r * 1024 + o]);
}

// ---------- mixture: logits + soft decision tree ----------
__global__ void __launch_bounds__(128) mixture_kernel(
    const float* __restrict__ x, const float* __restrict__ nw,
    const float* __restrict__ nb, float* __restrict__ mixture) {
  __shared__ float xrow[D_IN];
  __shared__ float lg[128];
  const int b = blockIdx.x;
  const int t = threadIdx.x;
  const float* xr = x + (size_t)b * D_IN;
  for (int i = t; i < D_IN; i += blockDim.x) xrow[i] = xr[i];
  __syncthreads();
  if (t < N_NODES) {
    const float* w = nw + (size_t)t * D_IN;
    float acc = 0.f;
    for (int i = 0; i < D_IN; i += 4) {
      acc += xrow[i] * w[i] + xrow[i + 1] * w[i + 1]
           + xrow[i + 2] * w[i + 2] + xrow[i + 3] * w[i + 3];
    }
    lg[t] = acc + nb[t];
  }
  __syncthreads();
  float prod = 1.f;
#pragma unroll
  for (int d = 0; d < 7; ++d) {
    int node = (1 << d) - 1 + (t >> (7 - d));
    int bit = (t >> (6 - d)) & 1;
    float p = 1.f / (1.f + __expf(-lg[node]));
    prod *= bit ? p : (1.f - p);
  }
  mixture[(size_t)b * N_LEAVES + t] = prod;
}

// ---------- GEMM1: G[b, l*64+j] = bf16( relu(x@w1s[l]+b1s[l]) * mixture[b,l] )
// block: 128 rows x 64 cols (one leaf), 8 waves; K-step 64; double-buffered LDS pipeline.
__global__ void __launch_bounds__(256) gemm1_kernel(
    const unsigned short* __restrict__ xb,
    const unsigned short* __restrict__ w1t,   // [(l*64+j)][1024]
    const float* __restrict__ b1s,
    const float* __restrict__ mixture,
    unsigned short* __restrict__ G) {
  __shared__ __align__(16) unsigned short As[2][128 * KSTEP];   // 2 x 16 KiB
  __shared__ __align__(16) unsigned short Bts[2][64 * KSTEP];   // 2 x  8 KiB
  __shared__ float mixRow[128];

  const int rowbase = blockIdx.x * 128;
  const int l = blockIdx.y;
  const int t = threadIdx.x;
  const int wave = t >> 5, lane = t & 31, half = lane >> 4, nlo = lane & 15;

  if (t < 128) mixRow[t] = mixture[(size_t)(rowbase + t) * N_LEAVES + l];

  // staging slices: A = 64B/thread (4x b128), B = 32B/thread (2x b128)
  const int arow = t >> 1, ahalf = t & 1;
  const unsigned short* aSrc = &xb[(size_t)(rowbase + arow) * D_IN + ahalf * 32];
  const int aOff = arow * KSTEP + ahalf * 32;
  const int bj = t >> 2, bq = t & 3;
  const unsigned short* bSrc = &w1t[(size_t)(l * LEAF + bj) * D_IN + bq * 16];
  const int bOff = bj * KSTEP + bq * 16;

  v8f z = {0.f, 0.f, 0.f, 0.f, 0.f, 0.f, 0.f, 0.f};
  v8f accs[4] = {z, z, z, z};

  const int NT = D_IN / KSTEP;   // 16

  // ---- prologue: stage tile 0 into buffer 0 ----
#if USE_ASYNC
#pragma unroll
  for (int c = 0; c < 4; ++c)
    __builtin_amdgcn_global_load_async_to_lds_b128(glb_cast(aSrc + c * 8),
                                                   lds_cast(&As[0][aOff + c * 8]), 0, 0);
#pragma unroll
  for (int c = 0; c < 2; ++c)
    __builtin_amdgcn_global_load_async_to_lds_b128(glb_cast(bSrc + c * 8),
                                                   lds_cast(&Bts[0][bOff + c * 8]), 0, 0);
  __builtin_amdgcn_s_wait_asynccnt(0);
#else
  {
    const uu4* s4 = (const uu4*)aSrc;
    uu4* d4 = (uu4*)&As[0][aOff];
    d4[0] = s4[0]; d4[1] = s4[1]; d4[2] = s4[2]; d4[3] = s4[3];
    const uu4* sb = (const uu4*)bSrc;
    uu4* db = (uu4*)&Bts[0][bOff];
    db[0] = sb[0]; db[1] = sb[1];
  }
#endif
  __syncthreads();

  for (int kt = 0; kt < NT; ++kt) {
    const int cur = kt & 1;
    const int nk0 = (kt + 1) * KSTEP;
    const bool more = (kt + 1 < NT);

#if USE_ASYNC
    if (more) {   // kick next tile's async copies; they overlap with the WMMAs below
#pragma unroll
      for (int c = 0; c < 4; ++c)
        __builtin_amdgcn_global_load_async_to_lds_b128(glb_cast(aSrc + nk0 + c * 8),
                                                       lds_cast(&As[cur ^ 1][aOff + c * 8]), 0, 0);
#pragma unroll
      for (int c = 0; c < 2; ++c)
        __builtin_amdgcn_global_load_async_to_lds_b128(glb_cast(bSrc + nk0 + c * 8),
                                                       lds_cast(&Bts[cur ^ 1][bOff + c * 8]), 0, 0);
    }
#else
    uu4 ra0, ra1, ra2, ra3, rb0, rb1;
    if (more) {   // issue global loads now; s_wait_loadcnt lands after the WMMAs
      const uu4* s4 = (const uu4*)(aSrc + nk0);
      ra0 = s4[0]; ra1 = s4[1]; ra2 = s4[2]; ra3 = s4[3];
      const uu4* sb = (const uu4*)(bSrc + nk0);
      rb0 = sb[0]; rb1 = sb[1];
    }
#endif

    // ---- compute on buffer `cur`: two K=32 chunks, 4 WMMAs each ----
#pragma unroll
    for (int c = 0; c < 2; ++c) {
      Frag16 a;
      const unsigned short* Ab = &As[cur][(wave * 16 + nlo) * KSTEP + c * 32];
      a.q[0] = *(const uu4*)&Ab[half * 8];
      a.q[1] = *(const uu4*)&Ab[16 + half * 8];
#pragma unroll
      for (int s = 0; s < 4; ++s) {
        Frag16 b;
        const unsigned short* Bb = &Bts[cur][(s * 16 + nlo) * KSTEP + c * 32];
        b.q[0] = *(const uu4*)&Bb[half * 8];
        b.q[1] = *(const uu4*)&Bb[16 + half * 8];
        accs[s] = wmma_bf16(a, b, accs[s]);
      }
    }

    if (more) {
#if USE_ASYNC
      __builtin_amdgcn_s_wait_asynccnt(0);
      __syncthreads();
#else
      uu4* d4 = (uu4*)&As[cur ^ 1][aOff];
      d4[0] = ra0; d4[1] = ra1; d4[2] = ra2; d4[3] = ra3;
      uu4* db = (uu4*)&Bts[cur ^ 1][bOff];
      db[0] = rb0; db[1] = rb1;
      __syncthreads();
#endif
    }
  }

  // ---- epilogue: +bias, relu, *mixture, bf16 store ----
#pragma unroll
  for (int s = 0; s < 4; ++s) {
    const int j = s * 16 + nlo;
    const float bb = b1s[l * LEAF + j];
    const int col = l * LEAF + j;
#pragma unroll
    for (int v = 0; v < 8; ++v) {
      const int m = half * 8 + v;
      float h = accs[s][v] + bb;
      h = fmaxf(h, 0.f) * mixRow[wave * 16 + m];
      G[(size_t)(rowbase + wave * 16 + m) * KFLAT + col] = f32_to_bf16(h);
    }
  }
}

// ---------- bias2: out = mixture @ b2s ----------
__global__ void __launch_bounds__(256) bias2_kernel(
    const float* __restrict__ mixture, const float* __restrict__ b2s,
    float* __restrict__ out) {
  const size_t idx = (size_t)blockIdx.x * 256 + threadIdx.x;
  const int b = (int)(idx >> 10);
  const int o = (int)(idx & 1023);
  const float* mrow = mixture + (size_t)b * N_LEAVES;
  float acc = 0.f;
#pragma unroll 4
  for (int l = 0; l < N_LEAVES; ++l) acc += mrow[l] * b2s[l * D_OUT + o];
  out[idx] = acc;
}

// ---------- GEMM2: out += G(8192x8192) @ W2flat(8192x1024), bf16 WMMA ----------
// block: 128 x 128, 8 waves x 8 subtiles; K-step 64; double-buffered LDS pipeline.
__global__ void __launch_bounds__(256) gemm2_kernel(
    const unsigned short* __restrict__ G,
    const unsigned short* __restrict__ w2t,   // [o][8192]
    float* __restrict__ out) {
  __shared__ __align__(16) unsigned short As[2][128 * KSTEP];   // 2 x 16 KiB
  __shared__ __align__(16) unsigned short Bts[2][128 * KSTEP];  // 2 x 16 KiB

  const int rowbase = blockIdx.x * 128;
  const int col0 = blockIdx.y * 128;
  const int t = threadIdx.x;
  const int wave = t >> 5, lane = t & 31, half = lane >> 4, nlo = lane & 15;

  // staging slices: 64B/thread each for A and B (4x b128)
  const int arow = t >> 1, ahalf = t & 1;
  const unsigned short* aSrc = &G[(size_t)(rowbase + arow) * KFLAT + ahalf * 32];
  const int aOff = arow * KSTEP + ahalf * 32;
  const int bn = t >> 1, bhalf = t & 1;
  const unsigned short* bSrc = &w2t[(size_t)(col0 + bn) * KFLAT + bhalf * 32];
  const int bOff = bn * KSTEP + bhalf * 32;

  v8f z = {0.f, 0.f, 0.f, 0.f, 0.f, 0.f, 0.f, 0.f};
  v8f acc[8] = {z, z, z, z, z, z, z, z};

  const int NT = KFLAT / KSTEP;   // 128

  // ---- prologue ----
#if USE_ASYNC
#pragma unroll
  for (int c = 0; c < 4; ++c) {
    __builtin_amdgcn_global_load_async_to_lds_b128(glb_cast(aSrc + c * 8),
                                                   lds_cast(&As[0][aOff + c * 8]), 0, 0);
    __builtin_amdgcn_global_load_async_to_lds_b128(glb_cast(bSrc + c * 8),
                                                   lds_cast(&Bts[0][bOff + c * 8]), 0, 0);
  }
  __builtin_amdgcn_s_wait_asynccnt(0);
#else
  {
    const uu4* s4 = (const uu4*)aSrc;
    uu4* d4 = (uu4*)&As[0][aOff];
    d4[0] = s4[0]; d4[1] = s4[1]; d4[2] = s4[2]; d4[3] = s4[3];
    const uu4* sb = (const uu4*)bSrc;
    uu4* db = (uu4*)&Bts[0][bOff];
    db[0] = sb[0]; db[1] = sb[1]; db[2] = sb[2]; db[3] = sb[3];
  }
#endif
  __syncthreads();

  for (int kt = 0; kt < NT; ++kt) {
    const int cur = kt & 1;
    const int nk0 = (kt + 1) * KSTEP;
    const bool more = (kt + 1 < NT);

#if USE_ASYNC
    if (more) {
#pragma unroll
      for (int c = 0; c < 4; ++c) {
        __builtin_amdgcn_global_load_async_to_lds_b128(glb_cast(aSrc + nk0 + c * 8),
                                                       lds_cast(&As[cur ^ 1][aOff + c * 8]), 0, 0);
        __builtin_amdgcn_global_load_async_to_lds_b128(glb_cast(bSrc + nk0 + c * 8),
                                                       lds_cast(&Bts[cur ^ 1][bOff + c * 8]), 0, 0);
      }
    }
#else
    uu4 ra0, ra1, ra2, ra3, rb0, rb1, rb2, rb3;
    if (more) {
      const uu4* s4 = (const uu4*)(aSrc + nk0);
      ra0 = s4[0]; ra1 = s4[1]; ra2 = s4[2]; ra3 = s4[3];
      const uu4* sb = (const uu4*)(bSrc + nk0);
      rb0 = sb[0]; rb1 = sb[1]; rb2 = sb[2]; rb3 = sb[3];
    }
#endif

    // ---- compute: two K=32 chunks x 8 WMMAs ----
#pragma unroll
    for (int c = 0; c < 2; ++c) {
      Frag16 a;
      const unsigned short* Ab = &As[cur][(wave * 16 + nlo) * KSTEP + c * 32];
      a.q[0] = *(const uu4*)&Ab[half * 8];
      a.q[1] = *(const uu4*)&Ab[16 + half * 8];
#pragma unroll
      for (int s = 0; s < 8; ++s) {
        Frag16 b;
        const unsigned short* Bb = &Bts[cur][(s * 16 + nlo) * KSTEP + c * 32];
        b.q[0] = *(const uu4*)&Bb[half * 8];
        b.q[1] = *(const uu4*)&Bb[16 + half * 8];
        acc[s] = wmma_bf16(a, b, acc[s]);
      }
    }

    if (more) {
#if USE_ASYNC
      __builtin_amdgcn_s_wait_asynccnt(0);
      __syncthreads();
#else
      uu4* d4 = (uu4*)&As[cur ^ 1][aOff];
      d4[0] = ra0; d4[1] = ra1; d4[2] = ra2; d4[3] = ra3;
      uu4* db = (uu4*)&Bts[cur ^ 1][bOff];
      db[0] = rb0; db[1] = rb1; db[2] = rb2; db[3] = rb3;
      __syncthreads();
#endif
    }
  }

  // ---- epilogue: accumulate into out (bias2 wrote the mixture@b2s term) ----
#pragma unroll
  for (int s = 0; s < 8; ++s) {
    const int col = col0 + s * 16 + nlo;
#pragma unroll
    for (int v = 0; v < 8; ++v) {
      const int row = rowbase + wave * 16 + half * 8 + v;
      out[(size_t)row * D_OUT + col] += acc[s][v];
    }
  }
}

// ---------- launcher ----------
extern "C" void kernel_launch(void* const* d_in, const int* in_sizes, int n_in,
                              void* d_out, int out_size, void* d_ws, size_t ws_size,
                              hipStream_t stream) {
  (void)in_sizes; (void)n_in; (void)out_size; (void)ws_size;
  const float* x   = (const float*)d_in[0];
  const float* nw  = (const float*)d_in[1];
  const float* nb  = (const float*)d_in[2];
  const float* w1s = (const float*)d_in[3];
  const float* b1s = (const float*)d_in[4];
  const float* w2s = (const float*)d_in[5];
  const float* b2s = (const float*)d_in[6];
  float* out = (float*)d_out;

  char* ws = (char*)d_ws;
  unsigned short* xb  = (unsigned short*)ws; ws += (size_t)B_TOT * D_IN * 2;            // 16 MiB
  unsigned short* w1t = (unsigned short*)ws; ws += (size_t)N_LEAVES * LEAF * D_IN * 2;  // 16 MiB
  unsigned short* w2t = (unsigned short*)ws; ws += (size_t)D_OUT * KFLAT * 2;           // 16 MiB
  float* mixture      = (float*)ws;          ws += (size_t)B_TOT * N_LEAVES * 4;        //  4 MiB
  unsigned short* G   = (unsigned short*)ws;                                            // 128 MiB

  conv_f32_bf16<<<4096, 256, 0, stream>>>(x, xb, B_TOT * D_IN);
  transpose_w1<<<(N_LEAVES * LEAF * D_IN) / 256, 256, 0, stream>>>(w1s, w1t);
  transpose_w2<<<(D_OUT * KFLAT) / 256, 256, 0, stream>>>(w2s, w2t);

  mixture_kernel<<<B_TOT, 128, 0, stream>>>(x, nw, nb, mixture);

  gemm1_kernel<<<dim3(B_TOT / 128, N_LEAVES), 256, 0, stream>>>(xb, w1t, b1s, mixture, G);

  bias2_kernel<<<(B_TOT * D_OUT) / 256, 256, 0, stream>>>(mixture, b2s, out);

  gemm2_kernel<<<dim3(B_TOT / 128, D_OUT / 128), 256, 0, stream>>>(G, w2t, out);
}
